// HybridConvTrans_80951543595136
// MI455X (gfx1250) — matrix operand order repaired and checked by reading, not compiled, not run on hardware
//
#include <hip/hip_runtime.h>
#include <hip/hip_bf16.h>

typedef __bf16 bf16;
typedef __attribute__((ext_vector_type(16))) __bf16 v16bf;
typedef __attribute__((ext_vector_type(8)))  float  v8f;
typedef __attribute__((ext_vector_type(4)))  unsigned int u32x4;

struct Frag16 { u32x4 a; u32x4 b; };   // 32 bytes -> v16bf

// ---- constants (reference shapes are fixed) ----
#define BB     8
#define NN     4096
#define CC     384
#define DD     64          // head dim == groups
#define MROWS  (BB*NN)     // 32768
#define NQKV   (3*CC)      // 1152
#define HH     64
#define WW     64

__device__ __forceinline__ bf16 f2bf(float f) {
    unsigned u = __builtin_bit_cast(unsigned, f);
    unsigned r = (u + 0x7FFFu + ((u >> 16) & 1u)) >> 16;
    unsigned short s = (unsigned short)r;
    return __builtin_bit_cast(bf16, s);
}
__device__ __forceinline__ float bf2f(bf16 h) {
    unsigned short s = __builtin_bit_cast(unsigned short, h);
    unsigned u = ((unsigned)s) << 16;
    return __builtin_bit_cast(float, u);
}

__device__ __forceinline__ void storev(float* p, float v) { *p = v; }
__device__ __forceinline__ void storev(bf16*  p, float v) { *p = f2bf(v); }

// ---------------- fp32 -> bf16 conversion ----------------
__global__ __launch_bounds__(256) void k_f32_to_bf16(const float* __restrict__ s,
                                                     bf16* __restrict__ d, int n) {
    int i = blockIdx.x * 256 + threadIdx.x;
    if (i < n) d[i] = f2bf(s[i]);
}

// ---------------- WMMA bf16 GEMM:  out[M,N] = A[M,K] @ W[N,K]^T + bias ----------------
// grid: (N/64, M/128), block 256 (8 waves). Wave w: rows m0..m0+15, cols n0..n0+63.
template <typename OutT>
__global__ __launch_bounds__(256)
void k_gemm_bf16(const bf16* __restrict__ A, const bf16* __restrict__ W,
                 const float* __restrict__ bias, OutT* __restrict__ out,
                 int M, int N, int K) {
    const int lane = threadIdx.x & 31;
    const int wave = threadIdx.x >> 5;
    const int l16  = lane & 15;
    const int hi   = lane >> 4;
    const int m0   = blockIdx.y * 128 + wave * 16;
    const int n0   = blockIdx.x * 64;

    const bf16* arow = A + (size_t)(m0 + l16) * K;
    const bf16* w0   = W + (size_t)(n0 +  0 + l16) * K;
    const bf16* w1   = W + (size_t)(n0 + 16 + l16) * K;
    const bf16* w2   = W + (size_t)(n0 + 32 + l16) * K;
    const bf16* w3   = W + (size_t)(n0 + 48 + l16) * K;

    v8f acc0 = {}, acc1 = {}, acc2 = {}, acc3 = {};

    // A frag (16x32): lane half hi holds K = hi*16 + 0..15 across the 8 vgprs
    // B frag (32x16): lane half hi holds K = hi*16 + 0..15 across the 8 vgprs
    for (int kk = 0; kk < K; kk += 32) {
        Frag16 fa;
        fa.a = *(const u32x4*)(arow + kk + hi * 16);
        fa.b = *(const u32x4*)(arow + kk + hi * 16 + 8);
        v16bf va = __builtin_bit_cast(v16bf, fa);

        Frag16 f0, f1, f2, f3;
        f0.a = *(const u32x4*)(w0 + kk + hi * 16);
        f0.b = *(const u32x4*)(w0 + kk + hi * 16 + 8);
        f1.a = *(const u32x4*)(w1 + kk + hi * 16);
        f1.b = *(const u32x4*)(w1 + kk + hi * 16 + 8);
        f2.a = *(const u32x4*)(w2 + kk + hi * 16);
        f2.b = *(const u32x4*)(w2 + kk + hi * 16 + 8);
        f3.a = *(const u32x4*)(w3 + kk + hi * 16);
        f3.b = *(const u32x4*)(w3 + kk + hi * 16 + 8);

        acc0 = __builtin_amdgcn_wmma_f32_16x16x32_bf16(false, va, false,
                 __builtin_bit_cast(v16bf, f0), (short)0, acc0, false, false);
        acc1 = __builtin_amdgcn_wmma_f32_16x16x32_bf16(false, va, false,
                 __builtin_bit_cast(v16bf, f1), (short)0, acc1, false, false);
        acc2 = __builtin_amdgcn_wmma_f32_16x16x32_bf16(false, va, false,
                 __builtin_bit_cast(v16bf, f2), (short)0, acc2, false, false);
        acc3 = __builtin_amdgcn_wmma_f32_16x16x32_bf16(false, va, false,
                 __builtin_bit_cast(v16bf, f3), (short)0, acc3, false, false);
    }

    const float b0 = bias[n0 +  0 + l16];
    const float b1 = bias[n0 + 16 + l16];
    const float b2 = bias[n0 + 32 + l16];
    const float b3 = bias[n0 + 48 + l16];

    // C/D layout: lane -> N = l16, vgpr e -> M = hi*8 + e
#pragma unroll
    for (int e = 0; e < 8; ++e) {
        const size_t ro = (size_t)(m0 + hi * 8 + e) * N;
        storev(out + ro + n0 +  0 + l16, acc0[e] + b0);
        storev(out + ro + n0 + 16 + l16, acc1[e] + b1);
        storev(out + ro + n0 + 32 + l16, acc2[e] + b2);
        storev(out + ro + n0 + 48 + l16, acc3[e] + b3);
    }
}

// ---------------- fc mix: F[b][g=dd][n][o] = sum_i fc_w[o,i]*Y[b, blk&4095, (blk>>12)*64+dd] + fc_b[o]
// blk = n*18 + i.  grid = MROWS/4 blocks, 256 threads (4 rows x 64 dd).
__global__ __launch_bounds__(256)
void k_fc_mix(const bf16* __restrict__ Y, const float* __restrict__ fc_w,
              const float* __restrict__ fc_b, float* __restrict__ F) {
    __shared__ float sw[162];
    __shared__ float sb[9];
    if (threadIdx.x < 162) sw[threadIdx.x] = fc_w[threadIdx.x];
    if (threadIdx.x < 9)   sb[threadIdx.x] = fc_b[threadIdx.x];
    __syncthreads();

    const int dd  = threadIdx.x & 63;
    const int idx = blockIdx.x * 4 + (threadIdx.x >> 6);
    const int b   = idx >> 12;
    const int n   = idx & 4095;

    float acc[9];
#pragma unroll
    for (int o = 0; o < 9; ++o) acc[o] = sb[o];

#pragma unroll
    for (int i = 0; i < 18; ++i) {
        const int blk = n * 18 + i;
        const int r   = blk & 4095;
        const int c0  = blk >> 12;
        const float v = bf2f(Y[((size_t)(b * 4096 + r)) * NQKV + c0 * 64 + dd]);
#pragma unroll
        for (int o = 0; o < 9; ++o) acc[o] += sw[o * 18 + i] * v;
    }
    float* dst = F + ((size_t)((b * 64 + dd) * 4096 + n)) * 9;
#pragma unroll
    for (int o = 0; o < 9; ++o) dst[o] = acc[o];
}

// ---------------- grouped 3x3 conv: per (b,g) tile 16x16, halo 18x18, 6 out ch / group
// grid: (16 tiles, 64 groups, 8 batch), block 256.
__global__ __launch_bounds__(256)
void k_dwconv(const float* __restrict__ F, const float* __restrict__ dep_w,
              const float* __restrict__ dep_b, float* __restrict__ O) {
    __shared__ float tile[18 * 18 * 9];
    __shared__ float wsm[486];
    __shared__ float bsm[6];

    const int g  = blockIdx.y;
    const int b  = blockIdx.z;
    const int th = (blockIdx.x >> 2) * 16;
    const int tw = (blockIdx.x & 3) * 16;

    for (int e = threadIdx.x; e < 486; e += 256)
        wsm[e] = dep_w[(size_t)(g * 6) * 81 + e];   // [c6][j][kh][kw] contiguous
    if (threadIdx.x < 6) bsm[threadIdx.x] = dep_b[g * 6 + threadIdx.x];

    const float* Fg = F + ((size_t)(b * 64 + g)) * 4096 * 9;
    for (int e = threadIdx.x; e < 18 * 18 * 9; e += 256) {
        const int r = e / 162, rem = e % 162, c = rem / 9, j = rem % 9;
        const int h = th - 1 + r, w = tw - 1 + c;
        float v = 0.f;
        if (h >= 0 && h < HH && w >= 0 && w < WW)
            v = Fg[(size_t)(h * WW + w) * 9 + j];
        tile[e] = v;
    }
    __syncthreads();

    const int lh = threadIdx.x >> 4, lw = threadIdx.x & 15;
    float acc[6];
#pragma unroll
    for (int c6 = 0; c6 < 6; ++c6) acc[c6] = bsm[c6];

#pragma unroll
    for (int kh = 0; kh < 3; ++kh)
#pragma unroll
        for (int kw = 0; kw < 3; ++kw) {
            const float* tp = &tile[((lh + kh) * 18 + (lw + kw)) * 9];
#pragma unroll
            for (int j = 0; j < 9; ++j) {
                const float v = tp[j];
#pragma unroll
                for (int c6 = 0; c6 < 6; ++c6)
                    acc[c6] += v * wsm[c6 * 81 + j * 9 + kh * 3 + kw];
            }
        }

    const int n = (th + lh) * WW + (tw + lw);
    float* op = O + ((size_t)(b * 4096 + n)) * CC + g * 6;
#pragma unroll
    for (int c6 = 0; c6 < 6; ++c6) op[c6] = acc[c6];
}

// ---------------- LayerNorm + exact GELU -> bf16 rows. grid = MROWS, block 128.
__global__ __launch_bounds__(128)
void k_ln_gelu(const float* __restrict__ O, const float* __restrict__ gma,
               const float* __restrict__ bta, bf16* __restrict__ Z) {
    __shared__ float s1[128];
    __shared__ float s2[128];
    const int row = blockIdx.x;
    const float* p = O + (size_t)row * CC;
    const int t = threadIdx.x;
    const float v0 = p[t], v1 = p[t + 128], v2 = p[t + 256];
    s1[t] = v0 + v1 + v2;
    s2[t] = v0 * v0 + v1 * v1 + v2 * v2;
    __syncthreads();
    for (int off = 64; off > 0; off >>= 1) {
        if (t < off) { s1[t] += s1[t + off]; s2[t] += s2[t + off]; }
        __syncthreads();
    }
    const float mu  = s1[0] * (1.f / 384.f);
    const float var = s2[0] * (1.f / 384.f) - mu * mu;
    const float rs  = rsqrtf(var + 1e-5f);

    bf16* z = Z + (size_t)row * CC;
    float vv[3] = { v0, v1, v2 };
#pragma unroll
    for (int k = 0; k < 3; ++k) {
        const int c = t + k * 128;
        const float y  = (vv[k] - mu) * rs * gma[c] + bta[c];
        const float ge = 0.5f * y * (1.f + erff(y * 0.70710678118654752f));
        z[c] = f2bf(ge);
    }
}

static inline size_t align256(size_t x) { return (x + 255) & ~(size_t)255; }

extern "C" void kernel_launch(void* const* d_in, const int* in_sizes, int n_in,
                              void* d_out, int out_size, void* d_ws, size_t ws_size,
                              hipStream_t stream) {
    const float* x      = (const float*)d_in[0];
    const float* q_w    = (const float*)d_in[1];
    const float* q_b    = (const float*)d_in[2];
    const float* kv_w   = (const float*)d_in[3];
    const float* kv_b   = (const float*)d_in[4];
    const float* fc_w   = (const float*)d_in[5];
    const float* fc_b   = (const float*)d_in[6];
    const float* dep_w  = (const float*)d_in[7];
    const float* dep_b  = (const float*)d_in[8];
    const float* ln_g   = (const float*)d_in[9];
    const float* ln_b   = (const float*)d_in[10];
    const float* proj_w = (const float*)d_in[11];
    const float* proj_b = (const float*)d_in[12];
    float* out = (float*)d_out;

    // ---- workspace layout ----
    char* ws = (char*)d_ws;
    size_t off = 0;
    bf16* xb    = (bf16*)(ws + off); off = align256(off + (size_t)MROWS * CC * 2);   // also Z later
    bf16* wqkv  = (bf16*)(ws + off); off = align256(off + (size_t)NQKV * CC * 2);
    bf16* projw = (bf16*)(ws + off); off = align256(off + (size_t)CC * CC * 2);
    float* biasq = (float*)(ws + off); off = align256(off + (size_t)NQKV * 4);
    bf16* Y     = (bf16*)(ws + off); off = align256(off + (size_t)MROWS * NQKV * 2); // also O later
    float* F    = (float*)(ws + off); off = align256(off + (size_t)BB * DD * NN * 9 * 4);
    float* O    = (float*)Y;     // Y dead after fc_mix; O is 50MB < 75MB
    bf16*  Z    = xb;            // xb dead after GEMM1

    // 1) convert inputs to bf16
    {
        int n = MROWS * CC;
        k_f32_to_bf16<<<(n + 255) / 256, 256, 0, stream>>>(x, xb, n);
        n = CC * CC;
        k_f32_to_bf16<<<(n + 255) / 256, 256, 0, stream>>>(q_w, wqkv, n);
        n = 2 * CC * CC;
        k_f32_to_bf16<<<(n + 255) / 256, 256, 0, stream>>>(kv_w, wqkv + (size_t)CC * CC, n);
        n = CC * CC;
        k_f32_to_bf16<<<(n + 255) / 256, 256, 0, stream>>>(proj_w, projw, n);
    }
    // concat biases [q_b ; kv_b]
    hipMemcpyAsync(biasq,        q_b,  (size_t)CC * 4,     hipMemcpyDeviceToDevice, stream);
    hipMemcpyAsync(biasq + CC,   kv_b, (size_t)2 * CC * 4, hipMemcpyDeviceToDevice, stream);

    // 2) fused QKV GEMM: Y[32768,1152] bf16
    {
        dim3 grid(NQKV / 64, MROWS / 128);
        k_gemm_bf16<bf16><<<grid, 256, 0, stream>>>(xb, wqkv, biasq, Y, MROWS, NQKV, CC);
    }
    // 3) fc mix -> F[b][g][n][9]
    k_fc_mix<<<MROWS / 4, 256, 0, stream>>>(Y, fc_w, fc_b, F);
    // 4) grouped 3x3 conv -> O[b][n][384]
    {
        dim3 grid(16, DD, BB);
        k_dwconv<<<grid, 256, 0, stream>>>(F, dep_w, dep_b, O);
    }
    // 5) LayerNorm + GELU -> Z bf16
    k_ln_gelu<<<MROWS, 128, 0, stream>>>(O, ln_g, ln_b, Z);
    // 6) projection GEMM -> d_out fp32
    {
        dim3 grid(CC / 64, MROWS / 128);
        k_gemm_bf16<float><<<grid, 256, 0, stream>>>(Z, projw, proj_b, out, MROWS, CC, CC);
    }
}
